// LearnableIPRMPNN_89876485636290
// MI455X (gfx1250) — matrix operand order repaired and checked by reading, not compiled, not run on hardware
//
#include <hip/hip_runtime.h>

typedef __attribute__((ext_vector_type(16))) _Float16 v16h;
typedef __attribute__((ext_vector_type(8)))  _Float16 v8h;
typedef __attribute__((ext_vector_type(8)))  float    v8f;

namespace {
constexpr int NG    = 4096;  // graphs
constexpr int NPG   = 64;    // nodes per graph
constexpr int IN_D  = 128;
constexpr int H_D   = 128;
constexpr int NV    = 16;    // virtual nodes
constexpr int KTOP  = 8;
constexpr int OUTD  = 10;
constexpr int NTHREADS = 256;  // 8 waves (wave32)

// packed f16 weight layout in d_ws (element offsets, _Float16 units)
constexpr int WS_WE = 0;                    // W_emb^T [n][k] 128x128
constexpr int WS_V1 = 16384;                // vW1^T
constexpr int WS_V2 = 32768;                // vW2^T
constexpr int WS_AT = 49152;                // A^T [v][k] 16x128
constexpr int WS_HALFS = 51200;             // total halfs (102400 bytes)
}

// ---- WMMA fragment loaders (CDNA5 16x16x32 f16, ISA 7.12.2 layouts) ----
// A 16x32 (MxK): lanes 0-15 row M=lane&15, K = k0+{0..7,16..23}; lanes 16-31 K +8
__device__ __forceinline__ v16h frag_a(const _Float16* base, int ld, int m0, int k0, int lane) {
  const _Float16* p = base + (m0 + (lane & 15)) * ld + k0 + ((lane >> 4) << 3);
  union { v16h v; v8h h[2]; } u;
  u.h[0] = *(const v8h*)(p);
  u.h[1] = *(const v8h*)(p + 16);
  return u.v;
}
// B 32x16 (KxN), stored transposed [N][K]: lanes 0-15 col N=lane&15, K = k0+{0..7,16..23};
// lanes 16-31: K +8
__device__ __forceinline__ v16h frag_b(const _Float16* baseT, int ld, int n0, int k0, int lane) {
  const _Float16* p = baseT + (n0 + (lane & 15)) * ld + k0 + ((lane >> 4) << 3);
  union { v16h v; v8h h[2]; } u;
  u.h[0] = *(const v8h*)(p);
  u.h[1] = *(const v8h*)(p + 16);
  return u.v;
}

__device__ __forceinline__ v8f wmma_f16(v16h a, v16h b, v8f c) {
  return __builtin_amdgcn_wmma_f32_16x16x32_f16(false, a, false, b, (short)0, c, false, false);
}

// ---- one-time weight pack: f32 row-major -> f16 transposed in d_ws ----
__global__ __launch_bounds__(256)
void pack_weights(const float* __restrict__ Wemb, const float* __restrict__ vW1,
                  const float* __restrict__ vW2, const float* __restrict__ Amat,
                  _Float16* __restrict__ ws)
{
  const int idx = blockIdx.x * 256 + threadIdx.x;
  const int stride = gridDim.x * 256;
  for (int j = idx; j < H_D * IN_D; j += stride) {
    int n = j >> 7, k = j & 127;                 // output [n][k], coalesced writes
    ws[WS_WE + j] = (_Float16)Wemb[k * H_D + n];
    ws[WS_V1 + j] = (_Float16)vW1[k * H_D + n];
    ws[WS_V2 + j] = (_Float16)vW2[k * H_D + n];
  }
  for (int j = idx; j < NV * H_D; j += stride) {
    int v = j >> 7, k = j & 127;                 // A is (1, H, V)
    ws[WS_AT + j] = (_Float16)Amat[k * NV + v];
  }
}

__global__ __launch_bounds__(NTHREADS)
void iprmpnn_fused(const float* __restrict__ x,
                   const float* __restrict__ Wemb,
                   const float* __restrict__ bemb,
                   const float* __restrict__ Amat,
                   const float* __restrict__ vW1, const float* __restrict__ vb1,
                   const float* __restrict__ vW2, const float* __restrict__ vb2,
                   const float* __restrict__ mW1, const float* __restrict__ mb1,
                   const float* __restrict__ mW2, const float* __restrict__ mb2,
                   const _Float16* __restrict__ ws, int packed,
                   float* __restrict__ out)
{
  // Persistent LDS (B operands stored transposed [N][K] for contiguous frag_b).
  __shared__ __align__(32) _Float16 sWe[H_D][IN_D];  // W_emb^T   32 KB
  __shared__ __align__(32) _Float16 sAt[NV][H_D];    // A^T        4 KB
  __shared__ __align__(32) _Float16 sV1[H_D][H_D];   // vW1^T     32 KB
  __shared__ __align__(32) _Float16 sV2[H_D][H_D];   // vW2^T     32 KB
  __shared__ __align__(32) _Float16 sEt[H_D][NPG];   // emb^T     16 KB
  __shared__ __align__(32) _Float16 sWv[NV][NPG];    // softmax w  2 KB
  // Phase-disjoint overlays (16 KB each):
  __shared__ __align__(32) unsigned char sOvl1[16384]; // {x tile | aff/val/vn2/gf/hf}
  __shared__ __align__(32) unsigned char sOvl2[16384]; // {emb rm | vn + h1}

  _Float16 (*sX)[IN_D] = (_Float16(*)[IN_D])sOvl1;        // phases 0-1
  float    (*sAff)[NPG] = (float(*)[NPG])sOvl1;           // phases 2-3
  float    (*sVal)[NPG] = (float(*)[NPG])(sOvl1 + 4096);  // phases 3-4
  float    (*sVn2)[H_D] = (float(*)[H_D])(sOvl1 + 8192);  // phases 7-8
  float*   sGf = (float*)(sOvl1 + 4096);                  // phases 8-9 (sVal dead)
  float*   sHf = (float*)(sOvl1 + 4608);                  // phases 9-10

  _Float16 (*sE)[H_D]  = (_Float16(*)[H_D])sOvl2;         // phases 1-2
  _Float16 (*sVn)[H_D] = (_Float16(*)[H_D])sOvl2;         // phases 5-6
  _Float16 (*sH1)[H_D] = (_Float16(*)[H_D])(sOvl2 + 8192);// phases 6-7

  const int tid  = threadIdx.x;
  const int lane = tid & 31;
  const int wave = tid >> 5;
  const int g    = blockIdx.x;

  // ---- phase 0: stage weights + x tile ----
  if (packed) {
    // pure vectorized copies of pre-packed f16 weights
    const uint4* swe = (const uint4*)(ws + WS_WE);
    const uint4* sv1 = (const uint4*)(ws + WS_V1);
    const uint4* sv2 = (const uint4*)(ws + WS_V2);
    const uint4* sat = (const uint4*)(ws + WS_AT);
    uint4* dwe = (uint4*)&sWe[0][0];
    uint4* dv1 = (uint4*)&sV1[0][0];
    uint4* dv2 = (uint4*)&sV2[0][0];
    uint4* dat = (uint4*)&sAt[0][0];
    for (int i = tid; i < (H_D * IN_D) / 8; i += NTHREADS) {
      dwe[i] = swe[i];
      dv1[i] = sv1[i];
      dv2[i] = sv2[i];
    }
    for (int i = tid; i < (NV * H_D) / 8; i += NTHREADS) dat[i] = sat[i];
  } else {
    // fallback: convert f32 -> f16 transposed in-kernel
    for (int i = tid; i < IN_D * H_D; i += NTHREADS) {
      int k = i >> 7, n = i & 127;
      sWe[n][k] = (_Float16)Wemb[i];
      sV1[n][k] = (_Float16)vW1[i];
      sV2[n][k] = (_Float16)vW2[i];
    }
    for (int i = tid; i < H_D * NV; i += NTHREADS) {
      int k = i >> 4, v = i & 15;
      sAt[v][k] = (_Float16)Amat[i];
    }
  }
  {
    const float4* x4 = (const float4*)(x + (size_t)g * NPG * IN_D);
    uint2* dst = (uint2*)&sX[0][0];
    for (int i = tid; i < (NPG * IN_D) / 4; i += NTHREADS) {
      float4 f = x4[i];
      union { _Float16 h[4]; uint2 u; } p;
      p.h[0] = (_Float16)f.x; p.h[1] = (_Float16)f.y;
      p.h[2] = (_Float16)f.z; p.h[3] = (_Float16)f.w;
      dst[i] = p.u;
    }
  }
  __syncthreads();

  // ---- phase 1: emb = x @ W_emb + b_emb : [64 x 128], K=128 ----
  for (int t = wave; t < 32; t += 8) {     // 4 M-tiles x 8 N-tiles (nt const/wave)
    int mt = t >> 3, nt = t & 7;
    v8f c = {};
#pragma unroll
    for (int k0 = 0; k0 < IN_D; k0 += 32) {
      v16h a = frag_a(&sX[0][0], IN_D, mt * 16, k0, lane);
      v16h b = frag_b(&sWe[0][0], IN_D, nt * 16, k0, lane);
      c = wmma_f16(a, b, c);
    }
    int n  = nt * 16 + (lane & 15);
    int mb = mt * 16 + ((lane & 16) >> 1); // accumulator: VGPR j -> M = mb + j
    float bias = bemb[n];
#pragma unroll
    for (int j = 0; j < 8; ++j) {
      _Float16 e = (_Float16)(c[j] + bias);
      sE[mb + j][n]  = e;                  // row-major (A operand for aff)
      sEt[n][mb + j] = e;                  // transposed (B operand for pooling)
    }
  }
  __syncthreads();

  // ---- phase 2: aff[v][n] = (emb @ A)[n][v], stored transposed : [16 x 64] ----
  if (wave < 4) {
    int mt = wave;
    v8f c = {};
#pragma unroll
    for (int k0 = 0; k0 < H_D; k0 += 32) {
      v16h a = frag_a(&sE[0][0], H_D, mt * 16, k0, lane);
      v16h b = frag_b(&sAt[0][0], H_D, 0, k0, lane);
      c = wmma_f16(a, b, c);
    }
    int v  = lane & 15;
    int mb = mt * 16 + ((lane & 16) >> 1);
#pragma unroll
    for (int j = 0; j < 8; ++j) sAff[v][mb + j] = c[j];
  }
  __syncthreads();

  // ---- phase 3: top-k mask (rank counting, stable like lax.top_k) ----
  for (int i = tid; i < NV * NPG; i += NTHREADS) {
    int v = i >> 6, n = i & 63;
    float an = sAff[v][n];
    int rank = 0;
    for (int m = 0; m < NPG; ++m) {
      float am = sAff[v][m];
      rank += (am > an) || ((am == an) && (m < n));
    }
    // faithful: non-top-k entries zeroed (NOT -inf) before softmax
    sVal[v][n] = (rank < KTOP) ? an : 0.0f;
  }
  __syncthreads();

  // ---- phase 4: softmax over nodes, per virtual node ----
  if (tid < NV) {
    int v = tid;
    float mx = sVal[v][0];
    for (int n = 1; n < NPG; ++n) mx = fmaxf(mx, sVal[v][n]);
    float s = 0.f;
    for (int n = 0; n < NPG; ++n) {
      float e = __expf(sVal[v][n] - mx);
      sVal[v][n] = e;                      // cache exp, reuse below
      s += e;
    }
    float inv = 1.0f / s;
    for (int n = 0; n < NPG; ++n)
      sWv[v][n] = (_Float16)(sVal[v][n] * inv);
  }
  __syncthreads();

  // ---- phase 5: vn = w @ emb : [16 x 128], K=64 (one N-tile per wave) ----
  {
    int nt = wave;
    v8f c = {};
#pragma unroll
    for (int k0 = 0; k0 < NPG; k0 += 32) {
      v16h a = frag_a(&sWv[0][0], NPG, 0, k0, lane);
      v16h b = frag_b(&sEt[0][0], NPG, nt * 16, k0, lane);
      c = wmma_f16(a, b, c);
    }
    int n  = nt * 16 + (lane & 15);
    int mb = (lane & 16) >> 1;
#pragma unroll
    for (int j = 0; j < 8; ++j) sVn[mb + j][n] = (_Float16)c[j];
  }
  __syncthreads();

  // ---- phase 6: h1 = relu(vn @ vW1 + vb1) ----
  {
    int nt = wave;
    v8f c = {};
#pragma unroll
    for (int k0 = 0; k0 < H_D; k0 += 32) {
      v16h a = frag_a(&sVn[0][0], H_D, 0, k0, lane);
      v16h b = frag_b(&sV1[0][0], H_D, nt * 16, k0, lane);
      c = wmma_f16(a, b, c);
    }
    int n  = nt * 16 + (lane & 15);
    float bias = vb1[n];
    int mb = (lane & 16) >> 1;
#pragma unroll
    for (int j = 0; j < 8; ++j) {
      float h = c[j] + bias;
      sH1[mb + j][n] = (_Float16)(h > 0.f ? h : 0.f);
    }
  }
  __syncthreads();

  // ---- phase 7: vn2 = h1 @ vW2 + vb2 (f32 result) ----
  {
    int nt = wave;
    v8f c = {};
#pragma unroll
    for (int k0 = 0; k0 < H_D; k0 += 32) {
      v16h a = frag_a(&sH1[0][0], H_D, 0, k0, lane);
      v16h b = frag_b(&sV2[0][0], H_D, nt * 16, k0, lane);
      c = wmma_f16(a, b, c);
    }
    int n  = nt * 16 + (lane & 15);
    float bias = vb2[n];
    int mb = (lane & 16) >> 1;
#pragma unroll
    for (int j = 0; j < 8; ++j) sVn2[mb + j][n] = c[j] + bias;
  }
  __syncthreads();

  // ---- phase 8: gf = mean over virtual nodes ----
  if (tid < H_D) {
    float s = 0.f;
#pragma unroll
    for (int v = 0; v < NV; ++v) s += sVn2[v][tid];
    sGf[tid] = s * (1.0f / NV);
  }
  __syncthreads();

  // ---- phase 9: hf = relu(gf @ mW1 + mb1)  (tiny [1x128] GEMV, VALU f32) ----
  if (tid < H_D) {
    float s = mb1[tid];
    for (int k = 0; k < H_D; ++k) s = fmaf(sGf[k], mW1[k * H_D + tid], s);
    sHf[tid] = s > 0.f ? s : 0.f;
  }
  __syncthreads();

  // ---- phase 10: out = hf @ mW2 + mb2 ----
  if (tid < OUTD) {
    float s = mb2[tid];
    for (int k = 0; k < H_D; ++k) s = fmaf(sHf[k], mW2[k * OUTD + tid], s);
    out[(size_t)g * OUTD + tid] = s;
  }
}

extern "C" void kernel_launch(void* const* d_in, const int* in_sizes, int n_in,
                              void* d_out, int out_size, void* d_ws, size_t ws_size,
                              hipStream_t stream) {
  const float* x    = (const float*)d_in[0];
  const float* Wemb = (const float*)d_in[1];
  const float* bemb = (const float*)d_in[2];
  const float* Amat = (const float*)d_in[3];
  const float* vW1  = (const float*)d_in[4];
  const float* vb1  = (const float*)d_in[5];
  const float* vW2  = (const float*)d_in[6];
  const float* vb2  = (const float*)d_in[7];
  const float* mW1  = (const float*)d_in[8];
  const float* mb1  = (const float*)d_in[9];
  const float* mW2  = (const float*)d_in[10];
  const float* mb2  = (const float*)d_in[11];
  // d_in[12] edge_index, d_in[13] batch: unused by the forward pass
  float* out = (float*)d_out;

  const int packed = (ws_size >= (size_t)WS_HALFS * sizeof(_Float16)) ? 1 : 0;
  _Float16* ws = (_Float16*)d_ws;

  if (packed) {
    hipLaunchKernelGGL(pack_weights, dim3(64), dim3(256), 0, stream,
                       Wemb, vW1, vW2, Amat, ws);
  }
  hipLaunchKernelGGL(iprmpnn_fused, dim3(NG), dim3(NTHREADS), 0, stream,
                     x, Wemb, bemb, Amat, vW1, vb1, vW2, vb2,
                     mW1, mb1, mW2, mb2, ws, packed, out);
}
